// Quantizer_197568496138
// MI455X (gfx1250) — matrix operand order repaired and checked by reading, not compiled, not run on hardware
//
#include <hip/hip_runtime.h>

typedef __attribute__((ext_vector_type(16))) _Float16 v16h;
typedef __attribute__((ext_vector_type(8)))  float    v8f;

#define NROWS 32768
#define DDIM  256
#define KEMB  1024

static __device__ __forceinline__ v8f wmma_f16(v16h a, v16h b, v8f c) {
  // D = A(16x32 f16) * B(32x16 f16) + C(16x16 f32)
  return __builtin_amdgcn_wmma_f32_16x16x32_f16(false, a, false, b, (short)0, c, false, false);
}

static __device__ __forceinline__ v16h pack16(uint4 a, uint4 b) {
  union { uint4 q[2]; v16h v; } u;
  u.q[0] = a; u.q[1] = b;
  return u.v;
}

// ---------------------------------------------------------------------------
// Kernel 0: convert codebook W (fp32) -> hi/lo f16 pair, compute ||w||^2,
// zero the count histogram and loss partials.
// ---------------------------------------------------------------------------
__global__ __launch_bounds__(256) void vq_prep(const float* __restrict__ W,
                                               _Float16* __restrict__ Whi,
                                               _Float16* __restrict__ Wlo,
                                               float* __restrict__ wnorm,
                                               int* __restrict__ counts,
                                               float* __restrict__ partials) {
  const int r = blockIdx.x * blockDim.x + threadIdx.x;   // 0..1023
  if (r < KEMB) {
    float acc = 0.0f;
    const float* wr = W + (size_t)r * DDIM;
    _Float16* hr = Whi + (size_t)r * DDIM;
    _Float16* lr = Wlo + (size_t)r * DDIM;
    for (int d = 0; d < DDIM; ++d) {
      float w = wr[d];
      _Float16 h = (_Float16)w;
      hr[d] = h;
      lr[d] = (_Float16)(w - (float)h);
      acc += w * w;
    }
    wnorm[r] = acc;
    counts[r] = 0;
    partials[r] = 0.0f;
    partials[r + KEMB] = 0.0f;
  }
}

// ---------------------------------------------------------------------------
// Kernel 1: distances via hi/lo-split f16 WMMA with double-buffered
// async global->LDS staging of the B tiles; per-row argmin; gather.
// One wave = 16 rows; block = 8 waves = 128 rows; grid = 256 blocks.
// ---------------------------------------------------------------------------

// LDS tile layout (uint4 units): per buffer, per array (hi/lo):
// 16 code rows x (32 chunks + 1 pad) = 528 uint4.  hi at +0, lo at +528.
// Buffers at 0 and 1056.  Total 2112 uint4 = 33 KB.
#define ROWU4   33
#define ARRU4   (16 * ROWU4)   // 528
#define BUFU4   (2 * ARRU4)    // 1056

// Stage one 16-code tile (8KB hi + 8KB lo) into LDS buffer `buf`.
// Each of the 256 threads issues 4 async b128 copies -> ASYNCcnt += 4/wave.
static __device__ __forceinline__ void stage_tile(int tile, int buf, int tid,
                                                  unsigned lbase,
                                                  const _Float16* Whi,
                                                  const _Float16* Wlo) {
  const unsigned bufoff = lbase + (unsigned)(buf * BUFU4) * 16u;
#pragma unroll
  for (int r = 0; r < 2; ++r) {
    const int i = tid + r * 256;                    // chunk 0..511 (16B each)
    const int code = i >> 5;
    const int j = i & 31;
    const unsigned loff_hi = bufoff + (unsigned)(code * ROWU4 + j) * 16u;
    const unsigned loff_lo = loff_hi + (unsigned)ARRU4 * 16u;
    const unsigned long long ghi =
        (unsigned long long)(const char*)Whi + (size_t)tile * 8192 + (size_t)i * 16;
    const unsigned long long glo =
        (unsigned long long)(const char*)Wlo + (size_t)tile * 8192 + (size_t)i * 16;
    asm volatile("global_load_async_to_lds_b128 %0, %1, off"
                 :: "v"(loff_hi), "v"(ghi) : "memory");
    asm volatile("global_load_async_to_lds_b128 %0, %1, off"
                 :: "v"(loff_lo), "v"(glo) : "memory");
  }
}

__global__ __launch_bounds__(256) void vq_main(const float* __restrict__ x,
                                               const float* __restrict__ W,
                                               const _Float16* __restrict__ Whi,
                                               const _Float16* __restrict__ Wlo,
                                               const float* __restrict__ wnorm,
                                               int* __restrict__ idxout,
                                               int* __restrict__ counts,
                                               float* __restrict__ partials,
                                               float* __restrict__ qout) {
  __shared__ uint4 lds[2 * BUFU4];

  const int tid  = threadIdx.x;
  const int lane = tid & 31;
  const int wave = tid >> 5;
  const int rowbase = blockIdx.x * 128 + wave * 16;
  const int half = lane >> 4;     // which 16-lane half
  const int col  = lane & 15;     // A: row index;  B/C: column index

  // LDS byte offset of the shared buffer (low 32 bits of generic pointer).
  const unsigned lbase = (unsigned)(unsigned long long)(void*)&lds[0];

  // ---- Build resident A fragments (hi+lo) for this wave's 16 rows --------
  // A 16x32 f16 layout: lane(l<16) holds row l, K = kb+{0..7, 16..23};
  //                     lane(l>=16) holds row l-16, K = kb+{8..15, 24..31}.
  const float* xrow = x + (size_t)(rowbase + col) * DDIM;
  v16h ahi[8], alo[8];
#pragma unroll
  for (int kc = 0; kc < 8; ++kc) {
    const int o0 = kc * 32 + half * 8;
    const int o1 = o0 + 16;
    float4 x0 = *(const float4*)(xrow + o0);
    float4 x1 = *(const float4*)(xrow + o0 + 4);
    float4 x2 = *(const float4*)(xrow + o1);
    float4 x3 = *(const float4*)(xrow + o1 + 4);
    float v[8] = {x0.x, x0.y, x0.z, x0.w, x1.x, x1.y, x1.z, x1.w};
    float u[8] = {x2.x, x2.y, x2.z, x2.w, x3.x, x3.y, x3.z, x3.w};
#pragma unroll
    for (int i = 0; i < 8; ++i) {
      _Float16 h = (_Float16)v[i];
      ahi[kc][i] = h;
      alo[kc][i] = (_Float16)(v[i] - (float)h);
      _Float16 g = (_Float16)u[i];
      ahi[kc][8 + i] = g;
      alo[kc][8 + i] = (_Float16)(u[i] - (float)g);
    }
  }

  // Prologue: stage tiles 0 and 1 (ASYNCcnt per wave = 8).
  stage_tile(0, 0, tid, lbase, Whi, Wlo);
  stage_tile(1, 1, tid, lbase, Whi, Wlo);

  float minv[8];
  int   mini[8];
#pragma unroll
  for (int j = 0; j < 8; ++j) { minv[j] = 3.0e38f; mini[j] = 0; }

  // ---- Sweep the 64 code tiles (16 codes each) ---------------------------
  for (int t = 0; t < 64; ++t) {
    // Async loads complete in order: <=4 outstanding => tile t is resident.
    if (t < 63) asm volatile("s_wait_asynccnt 0x4" ::: "memory");
    else        asm volatile("s_wait_asynccnt 0x0" ::: "memory");
    __syncthreads();

    // B 32x16 f16 layout: lane(l<16) holds column l, K = kb+{0..7, 16..23};
    // lane(l>=16) holds column l-16, K = kb+{8..15, 24..31}.
    const uint4* bh = &lds[(t & 1) * BUFU4 + col * ROWU4 + half];
    const uint4* bl = bh + ARRU4;
    v8f c = {};
#pragma unroll
    for (int kc = 0; kc < 8; ++kc) {
      v16h bhi = pack16(bh[kc * 4], bh[kc * 4 + 2]);
      v16h blo = pack16(bl[kc * 4], bl[kc * 4 + 2]);
      c = wmma_f16(ahi[kc], bhi, c);   // x_hi . w_hi
      c = wmma_f16(alo[kc], bhi, c);   // x_lo . w_hi
      c = wmma_f16(ahi[kc], blo, c);   // x_hi . w_lo
    }

    const int code = t * 16 + col;
    const float wn = wnorm[code];
#pragma unroll
    for (int j = 0; j < 8; ++j) {
      float d = wn - 2.0f * c[j];      // ||x||^2 row-constant: dropped
      if (d < minv[j]) { minv[j] = d; mini[j] = code; }   // codes ascend
    }

    __syncthreads();                   // all waves done reading buffer t&1
    if (t + 2 < 64) stage_tile(t + 2, t & 1, tid, lbase, Whi, Wlo);
  }
  asm volatile("s_wait_asynccnt 0x0" ::: "memory");

  // ---- Cross-lane argmin within each 16-lane half ------------------------
  // C layout: slot j, lanes 0-15 -> row j; lanes 16-31 -> row j+8.
#pragma unroll
  for (int j = 0; j < 8; ++j) {
#pragma unroll
    for (int m = 1; m <= 8; m <<= 1) {
      float ov = __shfl_xor(minv[j], m, 32);
      int   oi = __shfl_xor(mini[j], m, 32);
      if (ov < minv[j] || (ov == minv[j] && oi < mini[j])) {
        minv[j] = ov; mini[j] = oi;
      }
    }
  }

  // ---- Gather quantized rows, squared error, indices, counts -------------
  float err = 0.0f;
#pragma unroll 1
  for (int r = 0; r < 16; ++r) {
    const int id = __shfl(mini[r & 7], (r >> 3) << 4, 32);
    const float* wr = W + (size_t)id * DDIM;
    const float* xr = x + (size_t)(rowbase + r) * DDIM;
    float*       qr = qout + (size_t)(rowbase + r) * DDIM;
    if (lane == 0) {
      idxout[rowbase + r] = id;
      atomicAdd(counts + id, 1);
    }
#pragma unroll
    for (int p = 0; p < 2; ++p) {
      const int o = lane * 4 + p * 128;
      float4 q  = *(const float4*)(wr + o);
      float4 xv = *(const float4*)(xr + o);
      *(float4*)(qr + o) = q;
      float e0 = q.x - xv.x, e1 = q.y - xv.y;
      float e2 = q.z - xv.z, e3 = q.w - xv.w;
      err += e0 * e0 + e1 * e1 + e2 * e2 + e3 * e3;
    }
  }
#pragma unroll
  for (int m = 1; m <= 16; m <<= 1) err += __shfl_xor(err, m, 32);
  if (lane == 0) partials[blockIdx.x * 8 + wave] = err;
}

// ---------------------------------------------------------------------------
// Kernel 2: one-hot encodings, single pass (zero + scatter fused).
// One block per row; encodings base is only 8B-aligned -> float2 stores.
// ---------------------------------------------------------------------------
__global__ __launch_bounds__(256) void vq_enc(const int* __restrict__ idx,
                                              float* __restrict__ enc) {
  const int row = blockIdx.x;
  const int s = idx[row];
  float* rp = enc + (size_t)row * KEMB;
  const int c0 = threadIdx.x * 2;
#pragma unroll
  for (int c = c0; c < KEMB; c += 512) {
    float2 v;
    v.x = (c     == s) ? 1.0f : 0.0f;
    v.y = (c + 1 == s) ? 1.0f : 0.0f;
    *(float2*)(rp + c) = v;
  }
}

// ---------------------------------------------------------------------------
// Kernel 3: deterministic final reduction -> latent_loss + perplexity.
// ---------------------------------------------------------------------------
__global__ __launch_bounds__(1024) void vq_final(const int* __restrict__ counts,
                                                 const float* __restrict__ partials,
                                                 float* __restrict__ loss,
                                                 float* __restrict__ perp) {
  __shared__ float s1[1024];
  __shared__ float s2[1024];
  const int t = threadIdx.x;
  float p = (float)counts[t] * (1.0f / (float)NROWS);
  s1[t] = p * logf(p + 1e-10f);
  s2[t] = partials[t] + partials[t + 1024];
  __syncthreads();
  for (int s = 512; s > 0; s >>= 1) {
    if (t < s) { s1[t] += s1[t + s]; s2[t] += s2[t + s]; }
    __syncthreads();
  }
  if (t == 0) {
    *perp = expf(-s1[0]);
    *loss = 1.25f * s2[0] * (1.0f / ((float)NROWS * (float)DDIM));
  }
}

// ---------------------------------------------------------------------------
extern "C" void kernel_launch(void* const* d_in, const int* in_sizes, int n_in,
                              void* d_out, int out_size, void* d_ws, size_t ws_size,
                              hipStream_t stream) {
  const float* f_emb = (const float*)d_in[0];
  const float* W     = (const float*)d_in[1];

  float* out  = (float*)d_out;
  float* qout = out;                               // [32768*256]
  float* loss = out + (size_t)NROWS * DDIM;        // scalar
  float* perp = loss + 1;                          // scalar
  float* enc  = perp + 1;                          // [32768*1024]

  char* ws = (char*)d_ws;
  _Float16* Whi   = (_Float16*)(ws);               // 512 KB
  _Float16* Wlo   = (_Float16*)(ws + 524288);      // 512 KB
  float* wnorm    = (float*)(ws + 1048576);        // 4 KB
  int*   idx      = (int*)(ws + 1052672);          // 128 KB
  int*   counts   = (int*)(ws + 1183744);          // 4 KB
  float* partials = (float*)(ws + 1187840);        // 8 KB (2048 floats)

  vq_prep<<<4, 256, 0, stream>>>(W, Whi, Wlo, wnorm, counts, partials);
  vq_main<<<256, 256, 0, stream>>>(f_emb, W, Whi, Wlo, wnorm, idx, counts,
                                   partials, qout);
  vq_enc<<<NROWS, 256, 0, stream>>>(idx, enc);
  vq_final<<<1, 1024, 0, stream>>>(counts, partials, loss, perp);
}
